// MultiHeadedAttention_55233279426863
// MI455X (gfx1250) — compile-verified, matrix operand order checked by reading, not run on hardware
//
#include <hip/hip_runtime.h>

// ---------------------------------------------------------------------------
// MHA for MI455X (gfx1250): bf16 WMMA (16x16x32), f32 accumulate,
// async global->LDS staging (ASYNCcnt) where access patterns allow.
// ---------------------------------------------------------------------------

typedef __attribute__((ext_vector_type(16))) __bf16 v16bf;
typedef __attribute__((ext_vector_type(8)))  float  v8f;

union AF { v16bf v; unsigned int u[8]; };   // A or B fragment (16 bf16)
union CF { v8f   v; float f[8]; };          // C/D fragment (8 f32)

__device__ __forceinline__ unsigned short f2bf(float x) {
  unsigned int u = __float_as_uint(x);
  u += 0x7FFFu + ((u >> 16) & 1u);          // round-to-nearest-even
  return (unsigned short)(u >> 16);
}

// K index of the r-th 32-bit pair of a 16-bit A/B fragment, per ISA layout:
// lanes 0-15 : VGPR0-3 -> K 0..7,  VGPR4-7 -> K 16..23
// lanes 16-31: VGPR0-3 -> K 8..15, VGPR4-7 -> K 24..31
__device__ __forceinline__ int kidx(int r, int g) {
  return ((r & 4) << 2) + g * 8 + (r & 3) * 2;
}

__device__ __forceinline__ void load_frag(AF& f, const unsigned short* rowptr, int g) {
#pragma unroll
  for (int r = 0; r < 8; ++r)
    f.u[r] = *(const unsigned int*)(rowptr + kidx(r, g));
}

__device__ __forceinline__ v8f wmma_bf16(const AF& a, const AF& b, const v8f& c) {
  return __builtin_amdgcn_wmma_f32_16x16x32_bf16(false, a.v, false, b.v,
                                                 (short)0, c, false, false);
}

// ---- CDNA5 async global->LDS copy: 16B per lane, tracked by ASYNCcnt ------
__device__ __forceinline__ void async_copy_b128(const void* gsrc, void* ldst) {
#if defined(__gfx1250__)
  // generic->LDS address = low 32 bits; generic->global address = identity
  asm volatile("global_load_async_to_lds_b128 %0, %1, off"
               :
               : "v"((unsigned)(unsigned long long)ldst),
                 "v"((unsigned long long)gsrc)
               : "memory");
#else
  *(uint4*)ldst = *(const uint4*)gsrc;
#endif
}

__device__ __forceinline__ void async_wait0() {
#if defined(__gfx1250__)
  asm volatile("s_wait_asynccnt 0x0" ::: "memory");
#endif
}

// ---------------------------------------------------------------------------
// f32 -> bf16 convert
// ---------------------------------------------------------------------------
__global__ void cvt_bf16_kernel(const float* __restrict__ s,
                                unsigned short* __restrict__ d, int n) {
  for (int i = blockIdx.x * blockDim.x + threadIdx.x; i < n;
       i += gridDim.x * blockDim.x)
    d[i] = f2bf(s[i]);
}

// ---------------------------------------------------------------------------
// GEMM: C[M,N] = A[M,K](bf16) * B[K,N](bf16) + bias ; out bf16 or f32
// 256 threads (8 waves), tile 128x128x32, wave tile 64x32,
// double-buffered LDS, async A-tile staging.
// ---------------------------------------------------------------------------
#define BM 128
#define BN 128
#define BK 32
#define TPAD 40   // padded row length (bf16): 80B stride, conflict-free, 16B aligned

template <int OUT_BF16>
__global__ __launch_bounds__(256)
void gemm_bias_kernel(const unsigned short* __restrict__ A,
                      const unsigned short* __restrict__ B,
                      const float* __restrict__ bias,
                      void* __restrict__ out,
                      int M, int N, int K) {
  __shared__ unsigned short At[2][BM * TPAD] __attribute__((aligned(16)));
  __shared__ unsigned short Bt[2][BN * TPAD] __attribute__((aligned(16)));  // [n][k]

  const int tid  = threadIdx.x;
  const int lane = tid & 31;
  const int w    = tid >> 5;
  const int g    = lane >> 4;
  const int c    = lane & 15;

  const int gridM = M / BM;
  const int bm = (blockIdx.x % gridM) * BM;
  const int bn = (blockIdx.x / gridM) * BN;
  const int mw = (w & 1) * 64;   // wave M offset (4 frags)
  const int nw = (w >> 1) * 32;  // wave N offset (2 frags)

  // A tile: 128x32, 16B per thread per phase, async DMA to LDS
  auto stageA = [&](int k0, int buf) {
#pragma unroll
    for (int ph = 0; ph < 2; ++ph) {
      int idx = tid * 8 + ph * 2048;
      int row = idx >> 5, kk = idx & 31;
      async_copy_b128(A + (size_t)(bm + row) * K + k0 + kk,
                      &At[buf][row * TPAD + kk]);
    }
  };
  // B tile: transpose scatter (coalesced global read, b16 LDS stores)
  auto stageB = [&](int k0, int buf) {
#pragma unroll
    for (int ph = 0; ph < 2; ++ph) {
      int idx = tid * 8 + ph * 2048;
      int kk = idx >> 7, n = idx & 127;
      uint4 wv = *(const uint4*)(B + (size_t)(k0 + kk) * N + bn + n);
      const unsigned short* ws = (const unsigned short*)&wv;
#pragma unroll
      for (int j = 0; j < 8; ++j) Bt[buf][(n + j) * TPAD + kk] = ws[j];
    }
  };

  CF acc[4][2];
#pragma unroll
  for (int mf = 0; mf < 4; ++mf)
#pragma unroll
    for (int nf = 0; nf < 2; ++nf)
#pragma unroll
      for (int r = 0; r < 8; ++r) acc[mf][nf].f[r] = 0.0f;

  const int nk = K / BK;
  stageA(0, 0);
  stageB(0, 0);

  for (int kt = 0; kt < nk; ++kt) {
    const int buf = kt & 1;
    async_wait0();
    __syncthreads();

    // 1) fragment loads for this tile (LDS in-order: these precede staging stores)
    AF af[4], bf[2];
#pragma unroll
    for (int mf = 0; mf < 4; ++mf)
      load_frag(af[mf], &At[buf][(mw + mf * 16 + c) * TPAD], g);
#pragma unroll
    for (int nf = 0; nf < 2; ++nf)
      load_frag(bf[nf], &Bt[buf][(nw + nf * 16 + c) * TPAD], g);

    // 2) kick off next tile staging (overlaps with WMMA below)
    if (kt + 1 < nk) {
      stageA(kt + 1, buf ^ 1);
      stageB(kt + 1, buf ^ 1);
    }

    // 3) matrix math
#pragma unroll
    for (int mf = 0; mf < 4; ++mf)
#pragma unroll
      for (int nf = 0; nf < 2; ++nf)
        acc[mf][nf].v = wmma_bf16(af[mf], bf[nf], acc[mf][nf].v);
  }

  // epilogue (compile-time output type)
#pragma unroll
  for (int mf = 0; mf < 4; ++mf)
#pragma unroll
    for (int nf = 0; nf < 2; ++nf) {
      int col = bn + nw + nf * 16 + c;
      float bv = bias[col];
#pragma unroll
      for (int r = 0; r < 8; ++r) {
        int row = bm + mw + mf * 16 + r + 8 * g;
        float val = acc[mf][nf].f[r] + bv;
        if (OUT_BF16)
          ((unsigned short*)out)[(size_t)row * N + col] = f2bf(val);
        else
          ((float*)out)[(size_t)row * N + col] = val;
      }
    }
}

// ---------------------------------------------------------------------------
// Attention: one WG (8 waves) = 256 queries of one (b,h). Two-pass flash:
// pass A = row max; pass B = exp-sum + P@V with WMMA. Key step = 32.
// K tile staged via async DMA; V tile transposed via scatter.
// ---------------------------------------------------------------------------
#define SEQ 2048
#define DMODEL 1024
#define KT_PAD 72   // 144B row stride (conflict-free, 16B aligned)
#define VT_PAD 34   // 68B row stride (conflict-free)
#define P_PAD  34

__global__ __launch_bounds__(256)
void attn_kernel(const unsigned short* __restrict__ Qb,
                 const unsigned short* __restrict__ Kb,
                 const unsigned short* __restrict__ Vb,
                 const long long* __restrict__ mask,
                 unsigned short* __restrict__ Ab) {
  __shared__ unsigned short Kt[32 * KT_PAD] __attribute__((aligned(16)));
  __shared__ unsigned short Vt[64 * VT_PAD] __attribute__((aligned(16)));
  __shared__ float maskadd[32];
  __shared__ unsigned short Pb[8][16 * P_PAD] __attribute__((aligned(16)));

  const int tid  = threadIdx.x;
  const int lane = tid & 31;
  const int w    = tid >> 5;
  const int g    = lane >> 4;
  const int c    = lane & 15;

  const int blk = blockIdx.x;          // b*128 + h*8 + qt
  const int qt = blk & 7;
  const int h  = (blk >> 3) & 15;
  const int b  = blk >> 7;

  const size_t rowbase = (size_t)b * SEQ;
  const int colbase = h * 64;
  const int q0 = qt * 256 + w * 32;    // this wave's 32 queries
  const float scale = 0.125f;          // 1/sqrt(64)

  // Q fragments: 2 M-frags x 2 k-steps (Dh=64), kept in registers
  AF qf[2][2];
#pragma unroll
  for (int mf = 0; mf < 2; ++mf)
#pragma unroll
    for (int ks = 0; ks < 2; ++ks) {
      const unsigned short* src =
          Qb + (rowbase + q0 + mf * 16 + c) * (size_t)DMODEL + colbase + ks * 32;
#pragma unroll
      for (int r = 0; r < 8; ++r)
        qf[mf][ks].u[r] = *(const unsigned int*)(src + kidx(r, g));
    }

  float mx[2][8];
#pragma unroll
  for (int mf = 0; mf < 2; ++mf)
#pragma unroll
    for (int r = 0; r < 8; ++r) mx[mf][r] = -1e30f;

  // ---------------- pass A: row max ----------------
  for (int kt = 0; kt < SEQ / 32; ++kt) {
    __syncthreads();
    {
      int idx = tid * 8, key = idx >> 6, d = idx & 63;
      async_copy_b128(Kb + (rowbase + kt * 32 + key) * (size_t)DMODEL + colbase + d,
                      Kt + key * KT_PAD + d);
      if (tid < 32) {
        long long mv = mask[(size_t)b * SEQ + kt * 32 + tid];
        maskadd[tid] = (mv == 0) ? -1e30f : 0.0f;
      }
    }
    async_wait0();
    __syncthreads();

    AF kf[2][2];
#pragma unroll
    for (int nf = 0; nf < 2; ++nf)
#pragma unroll
      for (int ks = 0; ks < 2; ++ks)
        load_frag(kf[nf][ks], Kt + (nf * 16 + c) * KT_PAD + ks * 32, g);

#pragma unroll
    for (int mf = 0; mf < 2; ++mf) {
      CF sc[2];
#pragma unroll
      for (int nf = 0; nf < 2; ++nf) {
#pragma unroll
        for (int r = 0; r < 8; ++r) sc[nf].f[r] = 0.0f;
#pragma unroll
        for (int ks = 0; ks < 2; ++ks)
          sc[nf].v = wmma_bf16(qf[mf][ks], kf[nf][ks], sc[nf].v);
        float ma = maskadd[nf * 16 + c];
#pragma unroll
        for (int r = 0; r < 8; ++r)
          mx[mf][r] = fmaxf(mx[mf][r], sc[nf].f[r] * scale + ma);
      }
    }
  }
  // reduce max across the 16 lanes of each half-group
#pragma unroll
  for (int mf = 0; mf < 2; ++mf)
#pragma unroll
    for (int r = 0; r < 8; ++r) {
      float v = mx[mf][r];
      v = fmaxf(v, __shfl_xor(v, 1, 32));
      v = fmaxf(v, __shfl_xor(v, 2, 32));
      v = fmaxf(v, __shfl_xor(v, 4, 32));
      v = fmaxf(v, __shfl_xor(v, 8, 32));
      mx[mf][r] = v;
    }

  // ---------------- pass B: exp-sum and P@V ----------------
  float l[2][8];
  CF acc[2][4];
#pragma unroll
  for (int mf = 0; mf < 2; ++mf)
#pragma unroll
    for (int r = 0; r < 8; ++r) l[mf][r] = 0.0f;
#pragma unroll
  for (int mf = 0; mf < 2; ++mf)
#pragma unroll
    for (int df = 0; df < 4; ++df)
#pragma unroll
      for (int r = 0; r < 8; ++r) acc[mf][df].f[r] = 0.0f;

  for (int kt = 0; kt < SEQ / 32; ++kt) {
    __syncthreads();
    {
      int idx = tid * 8, key = idx >> 6, d = idx & 63;
      async_copy_b128(Kb + (rowbase + kt * 32 + key) * (size_t)DMODEL + colbase + d,
                      Kt + key * KT_PAD + d);
      uint4 vv =
          *(const uint4*)(Vb + (rowbase + kt * 32 + key) * (size_t)DMODEL + colbase + d);
      const unsigned short* vs = (const unsigned short*)&vv;
#pragma unroll
      for (int j = 0; j < 8; ++j) Vt[(d + j) * VT_PAD + key] = vs[j];
      if (tid < 32) {
        long long mv = mask[(size_t)b * SEQ + kt * 32 + tid];
        maskadd[tid] = (mv == 0) ? -1e30f : 0.0f;
      }
    }
    async_wait0();
    __syncthreads();

    AF kf[2][2], vf[4];
#pragma unroll
    for (int nf = 0; nf < 2; ++nf)
#pragma unroll
      for (int ks = 0; ks < 2; ++ks)
        load_frag(kf[nf][ks], Kt + (nf * 16 + c) * KT_PAD + ks * 32, g);
#pragma unroll
    for (int df = 0; df < 4; ++df)
      load_frag(vf[df], Vt + (df * 16 + c) * VT_PAD, g);

#pragma unroll
    for (int mf = 0; mf < 2; ++mf) {
      CF sc[2];
#pragma unroll
      for (int nf = 0; nf < 2; ++nf) {
#pragma unroll
        for (int r = 0; r < 8; ++r) sc[nf].f[r] = 0.0f;
#pragma unroll
        for (int ks = 0; ks < 2; ++ks)
          sc[nf].v = wmma_bf16(qf[mf][ks], kf[nf][ks], sc[nf].v);
        float ma = maskadd[nf * 16 + c];
#pragma unroll
        for (int r = 0; r < 8; ++r) {
          float s = sc[nf].f[r] * scale + ma;
          float p = __expf(s - mx[mf][r]);
          l[mf][r] += p;
          Pb[w][(r + 8 * g) * P_PAD + nf * 16 + c] = f2bf(p);
        }
      }
      // re-load P in A-fragment layout (per-wave buffer, LDS is in-order)
      AF pf;
      load_frag(pf, &Pb[w][c * P_PAD], g);
#pragma unroll
      for (int df = 0; df < 4; ++df)
        acc[mf][df].v = wmma_bf16(pf, vf[df], acc[mf][df].v);
    }
  }

  // reduce row sums
#pragma unroll
  for (int mf = 0; mf < 2; ++mf)
#pragma unroll
    for (int r = 0; r < 8; ++r) {
      float v = l[mf][r];
      v += __shfl_xor(v, 1, 32);
      v += __shfl_xor(v, 2, 32);
      v += __shfl_xor(v, 4, 32);
      v += __shfl_xor(v, 8, 32);
      l[mf][r] = v;
    }

  // normalize + store attended (bf16, merged-head layout)
#pragma unroll
  for (int mf = 0; mf < 2; ++mf)
#pragma unroll
    for (int df = 0; df < 4; ++df) {
      int col = colbase + df * 16 + c;
#pragma unroll
      for (int r = 0; r < 8; ++r) {
        int q = q0 + mf * 16 + r + 8 * g;
        float val = acc[mf][df].f[r] / l[mf][r];
        Ab[(rowbase + q) * (size_t)DMODEL + col] = f2bf(val);
      }
    }
}

// ---------------------------------------------------------------------------
// launch
// ---------------------------------------------------------------------------
extern "C" void kernel_launch(void* const* d_in, const int* in_sizes, int n_in,
                              void* d_out, int out_size, void* d_ws, size_t ws_size,
                              hipStream_t stream) {
  (void)in_sizes; (void)n_in; (void)out_size; (void)ws_size;

  const float*     X    = (const float*)d_in[0];
  const long long* mask = (const long long*)d_in[1];
  const float*     WQ   = (const float*)d_in[2];
  const float*     WK   = (const float*)d_in[3];
  const float*     WV   = (const float*)d_in[4];
  const float*     WY   = (const float*)d_in[5];
  const float*     bQ   = (const float*)d_in[6];
  const float*     bK   = (const float*)d_in[7];
  const float*     bV   = (const float*)d_in[8];
  const float*     bY   = (const float*)d_in[9];
  float* Y = (float*)d_out;

  const int M = 4 * 2048;   // 8192 rows
  const int D = 1024;
  const int NX = M * D;     // 8388608
  const int NW = D * D;     // 1048576

  unsigned short* Xb  = (unsigned short*)d_ws;
  unsigned short* Qb  = Xb  + NX;
  unsigned short* Kb  = Qb  + NX;
  unsigned short* Vb  = Kb  + NX;
  unsigned short* Ab  = Vb  + NX;
  unsigned short* WQb = Ab  + NX;
  unsigned short* WKb = WQb + NW;
  unsigned short* WVb = WKb + NW;
  unsigned short* WYb = WVb + NW;

  cvt_bf16_kernel<<<2048, 256, 0, stream>>>(X,  Xb,  NX);
  cvt_bf16_kernel<<<512,  256, 0, stream>>>(WQ, WQb, NW);
  cvt_bf16_kernel<<<512,  256, 0, stream>>>(WK, WKb, NW);
  cvt_bf16_kernel<<<512,  256, 0, stream>>>(WV, WVb, NW);
  cvt_bf16_kernel<<<512,  256, 0, stream>>>(WY, WYb, NW);

  const int gemmBlocks = (M / BM) * (D / BN);  // 64 * 8 = 512
  gemm_bias_kernel<1><<<gemmBlocks, 256, 0, stream>>>(Xb, WQb, bQ, Qb, M, D, D);
  gemm_bias_kernel<1><<<gemmBlocks, 256, 0, stream>>>(Xb, WKb, bK, Kb, M, D, D);
  gemm_bias_kernel<1><<<gemmBlocks, 256, 0, stream>>>(Xb, WVb, bV, Vb, M, D, D);

  attn_kernel<<<4 * 16 * 8, 256, 0, stream>>>(Qb, Kb, Vb, mask, Ab);

  gemm_bias_kernel<0><<<gemmBlocks, 256, 0, stream>>>(Ab, WYb, bY, Y, M, D, D);
}
